// SelfAttention_17652315587449
// MI455X (gfx1250) — compile-verified, matrix operand order checked by reading, not run on hardware
//
#include <hip/hip_runtime.h>
#include <hip/hip_bf16.h>

#define SEQ 2048
#define DIM 4096
#define KV  1024
#define HD  128
#define NH  32

// Padded LDS row strides (in bf16 elements). Data rows are 32 (GEMM/VT/PS)
// or 128 (KT) elements; +8 elements (16B) of pad makes the dword stride
// 20 (or 68), and gcd(20,64)=gcd(68,64)=4 with 64/4=16, so the 16 lanes of a
// WMMA fragment load land in 16 distinct banks (64 banks x 4B on CDNA5).
#define ASTR 40    // 32-elem rows (GEMM A/B tiles, VT, PS)
#define KSTR 136   // 128-elem rows (flash K tile)

typedef __attribute__((ext_vector_type(16))) __bf16 v16bf;
typedef __attribute__((ext_vector_type(8)))  float  v8f;

union FragU { uint4 q[2]; v16bf v; };

__device__ __forceinline__ unsigned short f32_bf16(float f) {
  unsigned int u = __float_as_uint(f);
  return (unsigned short)((u + 0x7FFFu + ((u >> 16) & 1u)) >> 16);  // RNE
}

__device__ __forceinline__ unsigned int pack2(float a, float b) {
  return (unsigned int)f32_bf16(a) | ((unsigned int)f32_bf16(b) << 16);
}

// CDNA5 async global->LDS copy, 16B per lane. LDS offset is the low 32 bits
// of the generic pointer (LDS aperture: addr[31:0] is the LDS byte offset).
__device__ __forceinline__ void async_ld16(void* lds_dst, const void* gsrc) {
  const unsigned loff = (unsigned)(size_t)lds_dst;
  asm volatile("global_load_async_to_lds_b128 %0, %1, off"
               :: "v"(loff), "v"(gsrc) : "memory");
}
__device__ __forceinline__ void wait_async0() {
  asm volatile("s_wait_asynccnt 0x0" ::: "memory");
}

// Load one 16-bit A/B WMMA fragment from a row-major bf16 buffer.
// Caller points p at (row, k_base + half*8); elements 8..15 come from +16.
__device__ __forceinline__ v16bf frag_ld(const unsigned short* p) {
  FragU f;
  f.q[0] = *(const uint4*)(p);
  f.q[1] = *(const uint4*)(p + 16);
  return f.v;
}

__device__ __forceinline__ v8f wmma_bf16(v16bf a, v16bf b, v8f c) {
  return __builtin_amdgcn_wmma_f32_16x16x32_bf16(false, a, false, b, (short)0, c,
                                                 false, false);
}

// ---------------------------------------------------------------------------
// GEMM: C[m,n] = sum_k A[m,k] * B[n,k]   (B is a row-major f32 weight matrix)
// A: f32 (register-prefetch + convert) or bf16 (async global->LDS).
// C: f32, bf16, or bf16-transposed (Ct[n*ldc + m]) for the V projection.
// Block tile 128x128, BK=32, 256 threads = 8 waves (4x2), wave tile 32x64.
// Double-buffered LDS (bank-conflict-free padded rows), one barrier per
// K-step, next tile's loads in flight during the current tile's WMMAs.
// ---------------------------------------------------------------------------
template<bool A_BF16, bool C_BF16, bool C_TRANS>
__global__ __launch_bounds__(256) void gemm_wmma_kernel(
    const void* __restrict__ Ap, const float* __restrict__ B,
    void* __restrict__ Cp, int K, int ldc)
{
  __shared__ unsigned short As[2][128 * ASTR];
  __shared__ unsigned short Bs[2][128 * ASTR];

  const int tid  = threadIdx.x;
  const int lane = tid & 31;
  const int wave = tid >> 5;
  const int wm   = wave >> 1;      // 0..3
  const int wn   = wave & 1;       // 0..1
  const int half = lane >> 4;      // 0 or 1 (16-lane C-layout groups)
  const int l16  = lane & 15;

  const int bm = blockIdx.y * 128;
  const int bn = blockIdx.x * 128;

  // per-pass staging coordinates (2 passes x 8 elems per thread = 128x32 tile)
  int rowp[2], colp[2];
#pragma unroll
  for (int p = 0; p < 2; ++p) {
    const int e = (tid + p * 256) * 8;
    rowp[p] = e >> 5;
    colp[p] = e & 31;
  }

  v8f acc[2][4];
#pragma unroll
  for (int i = 0; i < 2; ++i)
#pragma unroll
    for (int j = 0; j < 4; ++j) { v8f z = {}; acc[i][j] = z; }

  // prefetch registers (raw f32; converted to bf16 at ds_store time)
  float4 ar0[2], ar1[2], br0[2], br1[2];

  const int nk = K >> 5;

  // prologue: loads for tile 0
#pragma unroll
  for (int p = 0; p < 2; ++p) {
    if constexpr (!A_BF16) {
      const float* s = (const float*)Ap + (size_t)(bm + rowp[p]) * K + colp[p];
      ar0[p] = *(const float4*)(s);
      ar1[p] = *(const float4*)(s + 4);
    }
    const float* sb = B + (size_t)(bn + rowp[p]) * K + colp[p];
    br0[p] = *(const float4*)(sb);
    br1[p] = *(const float4*)(sb + 4);
  }

  for (int kt = 0; kt < nk; ++kt) {
    const int k0  = kt << 5;
    const int buf = kt & 1;

    // ---- stage current tile into LDS[buf]
#pragma unroll
    for (int p = 0; p < 2; ++p) {
      if constexpr (A_BF16) {
        // bf16 A: raw async copy global->LDS, no VGPR round trip
        async_ld16(&As[buf][rowp[p] * ASTR + colp[p]],
                   (const unsigned short*)Ap +
                       (size_t)(bm + rowp[p]) * K + k0 + colp[p]);
      } else {
        uint4 d;
        d.x = pack2(ar0[p].x, ar0[p].y); d.y = pack2(ar0[p].z, ar0[p].w);
        d.z = pack2(ar1[p].x, ar1[p].y); d.w = pack2(ar1[p].z, ar1[p].w);
        *(uint4*)(&As[buf][rowp[p] * ASTR + colp[p]]) = d;
      }
      uint4 d;
      d.x = pack2(br0[p].x, br0[p].y); d.y = pack2(br0[p].z, br0[p].w);
      d.z = pack2(br1[p].x, br1[p].y); d.w = pack2(br1[p].z, br1[p].w);
      *(uint4*)(&Bs[buf][rowp[p] * ASTR + colp[p]]) = d;
    }

    // ---- issue global loads for next tile (in flight during compute)
    if (kt + 1 < nk) {
      const int kn = k0 + 32;
#pragma unroll
      for (int p = 0; p < 2; ++p) {
        if constexpr (!A_BF16) {
          const float* s = (const float*)Ap + (size_t)(bm + rowp[p]) * K + kn + colp[p];
          ar0[p] = *(const float4*)(s);
          ar1[p] = *(const float4*)(s + 4);
        }
        const float* sb = B + (size_t)(bn + rowp[p]) * K + kn + colp[p];
        br0[p] = *(const float4*)(sb);
        br1[p] = *(const float4*)(sb + 4);
      }
    }

    if constexpr (A_BF16) wait_async0();
    __syncthreads();

    // ---- fragment loads + 8 WMMAs from LDS[buf]
    v16bf af[2], bfr[4];
#pragma unroll
    for (int tm = 0; tm < 2; ++tm)
      af[tm] = frag_ld(&As[buf][(wm * 32 + tm * 16 + l16) * ASTR + half * 8]);
#pragma unroll
    for (int tn = 0; tn < 4; ++tn)
      bfr[tn] = frag_ld(&Bs[buf][(wn * 64 + tn * 16 + l16) * ASTR + half * 8]);
#pragma unroll
    for (int tm = 0; tm < 2; ++tm)
#pragma unroll
      for (int tn = 0; tn < 4; ++tn)
        acc[tm][tn] = wmma_bf16(af[tm], bfr[tn], acc[tm][tn]);
    // no trailing barrier: next iteration writes the *other* LDS buffer, and
    // its single barrier orders those writes against this iteration's reads.
  }

  // ---- epilogue (C layout: VGPR r -> row r + 8*half, lane%16 -> col)
#pragma unroll
  for (int tm = 0; tm < 2; ++tm) {
#pragma unroll
    for (int tn = 0; tn < 4; ++tn) {
      if constexpr (C_TRANS) {
        const int mbase = bm + wm * 32 + tm * 16 + half * 8;
        const int n     = bn + wn * 64 + tn * 16 + l16;
        uint4 d;
        d.x = pack2(acc[tm][tn][0], acc[tm][tn][1]);
        d.y = pack2(acc[tm][tn][2], acc[tm][tn][3]);
        d.z = pack2(acc[tm][tn][4], acc[tm][tn][5]);
        d.w = pack2(acc[tm][tn][6], acc[tm][tn][7]);
        unsigned short* Ct = (unsigned short*)Cp;
        *(uint4*)(Ct + (size_t)n * ldc + mbase) = d;
      } else if constexpr (C_BF16) {
        unsigned short* C = (unsigned short*)Cp;
        const int n = bn + wn * 64 + tn * 16 + l16;
#pragma unroll
        for (int r = 0; r < 8; ++r) {
          const int m = bm + wm * 32 + tm * 16 + r + half * 8;
          C[(size_t)m * ldc + n] = f32_bf16(acc[tm][tn][r]);
        }
      } else {
        float* C = (float*)Cp;
        const int n = bn + wn * 64 + tn * 16 + l16;
#pragma unroll
        for (int r = 0; r < 8; ++r) {
          const int m = bm + wm * 32 + tm * 16 + r + half * 8;
          C[(size_t)m * ldc + n] = acc[tm][tn][r];
        }
      }
    }
  }
}

// ---------------------------------------------------------------------------
// Flash-style causal GQA attention with async double-buffered K/V^T staging.
// grid = (SEQ/64, NH); block = 128 threads = 4 waves; wave owns 16 query rows.
// qb: bf16 [SEQ, DIM]   kb: bf16 [SEQ, KV]   vt: bf16 [KV, SEQ]   ob: bf16
// ---------------------------------------------------------------------------
__global__ __launch_bounds__(128) void flash_gqa_kernel(
    const unsigned short* __restrict__ qb,
    const unsigned short* __restrict__ kb,
    const unsigned short* __restrict__ vt,
    unsigned short* __restrict__ ob)
{
  __shared__ unsigned short KT[2][32 * KSTR];   // [key][d]   (padded rows)
  __shared__ unsigned short VT[2][128 * ASTR];  // [d][key]   (padded rows)
  __shared__ unsigned short PS[4][16 * ASTR];   // per-wave P (padded rows)

  const int tid  = threadIdx.x;
  const int lane = tid & 31;
  const int wave = tid >> 5;
  const int half = lane >> 4;
  const int l16  = lane & 15;

  const int h  = blockIdx.y;   // query head
  const int g  = h >> 2;       // kv group (N_REP = 4)
  const int m0 = blockIdx.x * 64 + wave * 16;

  // preload Q fragments (4 K-steps of 32 over HEAD_DIM)
  v16bf qf[4];
#pragma unroll
  for (int ks = 0; ks < 4; ++ks) {
    const unsigned short* p =
        qb + (size_t)(m0 + l16) * DIM + h * HD + ks * 32 + half * 8;
    FragU f;
    f.q[0] = *(const uint4*)(p);
    f.q[1] = *(const uint4*)(p + 16);
    qf[ks] = f.v;
  }

  float rmax[8], rsum[8];
  v8f o[8];
#pragma unroll
  for (int r = 0; r < 8; ++r) { rmax[r] = -3.0e38f; rsum[r] = 0.f; }
#pragma unroll
  for (int dt = 0; dt < 8; ++dt) { v8f z = {}; o[dt] = z; }

  const int nkb = (blockIdx.x + 1) * 2;      // 32-key blocks (causal limit)
  const float scale = 0.08838834764831845f;  // 1/sqrt(128)

  // async-stage one 32-key block (K tile + V^T tile) into buffer `buf`
  auto stage = [&](int ikb, int buf) {
    const int kb0 = ikb * 32;
#pragma unroll
    for (int p = 0; p < 4; ++p) {
      const int e = (tid + p * 128) * 8;
      const int trow = e >> 7, tcol = e & 127;
      async_ld16(&KT[buf][trow * KSTR + tcol],
                 kb + (size_t)(kb0 + trow) * KV + g * HD + tcol);
    }
#pragma unroll
    for (int p = 0; p < 4; ++p) {
      const int e = (tid + p * 128) * 8;
      const int drow = e >> 5, tcol = e & 31;
      async_ld16(&VT[buf][drow * ASTR + tcol],
                 vt + (size_t)(g * HD + drow) * SEQ + kb0 + tcol);
    }
  };

  stage(0, 0);
  wait_async0();
  __syncthreads();

  for (int ikb = 0; ikb < nkb; ++ikb) {
    const int buf = ikb & 1;
    const int kb0 = ikb * 32;

    // prefetch next key block into the other buffer (overlaps with compute;
    // all waves passed the barrier ending ikb-1, which last read that buffer)
    if (ikb + 1 < nkb) stage(ikb + 1, buf ^ 1);

    // S = Q K^T : 16 x 32 tile, two 16x16 n-tiles
    v8f s[2];
#pragma unroll
    for (int tn = 0; tn < 2; ++tn) {
      v8f a = {};
#pragma unroll
      for (int ks = 0; ks < 4; ++ks) {
        v16bf kf =
            frag_ld(&KT[buf][(tn * 16 + l16) * KSTR + ks * 32 + half * 8]);
        a = wmma_bf16(qf[ks], kf, a);
      }
      s[tn] = a;
    }

    // causal mask + scale
#pragma unroll
    for (int tn = 0; tn < 2; ++tn) {
      const int t = kb0 + tn * 16 + l16;
#pragma unroll
      for (int r = 0; r < 8; ++r) {
        const int m = m0 + r + half * 8;
        const float v = s[tn][r] * scale;
        s[tn][r] = (t > m) ? -3.0e38f : v;
      }
    }

    // online softmax per row (row r+8*half; 16-lane xor reductions, wave32)
#pragma unroll
    for (int r = 0; r < 8; ++r) {
      float vmax = fmaxf(s[0][r], s[1][r]);
#pragma unroll
      for (int off = 8; off >= 1; off >>= 1)
        vmax = fmaxf(vmax, __shfl_xor(vmax, off, 32));
      const float nmax  = fmaxf(rmax[r], vmax);
      const float alpha = __expf(rmax[r] - nmax);
      rmax[r] = nmax;
      const float p0 = __expf(s[0][r] - nmax);
      const float p1 = __expf(s[1][r] - nmax);
      float ps = p0 + p1;
#pragma unroll
      for (int off = 8; off >= 1; off >>= 1)
        ps += __shfl_xor(ps, off, 32);
      rsum[r] = rsum[r] * alpha + ps;
#pragma unroll
      for (int dt = 0; dt < 8; ++dt) o[dt][r] *= alpha;
      const int prow = r + half * 8;
      PS[wave][prow * ASTR + l16]      = f32_bf16(p0);
      PS[wave][prow * ASTR + 16 + l16] = f32_bf16(p1);
    }

    // O += P V : P (16x32) as A fragment via wave-local LDS (DS in-order/wave)
    v16bf pf = frag_ld(&PS[wave][l16 * ASTR + half * 8]);
#pragma unroll
    for (int dt = 0; dt < 8; ++dt) {
      v16bf vf = frag_ld(&VT[buf][(dt * 16 + l16) * ASTR + half * 8]);
      o[dt] = wmma_bf16(pf, vf, o[dt]);
    }

    if (ikb + 1 < nkb) wait_async0();  // next block's tiles landed in LDS
    __syncthreads();
  }

  // normalize + store bf16 attention output
#pragma unroll
  for (int dt = 0; dt < 8; ++dt) {
    const int col = h * HD + dt * 16 + l16;
#pragma unroll
    for (int r = 0; r < 8; ++r) {
      const int m = m0 + r + half * 8;
      ob[(size_t)m * DIM + col] = f32_bf16(o[dt][r] / rsum[r]);
    }
  }
}

// ---------------------------------------------------------------------------
extern "C" void kernel_launch(void* const* d_in, const int* in_sizes, int n_in,
                              void* d_out, int out_size, void* d_ws, size_t ws_size,
                              hipStream_t stream) {
  (void)in_sizes; (void)n_in; (void)out_size; (void)ws_size;
  const float* x  = (const float*)d_in[0];
  const float* Wq = (const float*)d_in[1];
  const float* Wk = (const float*)d_in[2];
  const float* Wv = (const float*)d_in[3];
  const float* Wo = (const float*)d_in[4];
  float* out = (float*)d_out;

  unsigned short* qb = (unsigned short*)d_ws;          // [SEQ, DIM]  bf16
  unsigned short* kb = qb + (size_t)SEQ * DIM;         // [SEQ, KV]   bf16
  unsigned short* vt = kb + (size_t)SEQ * KV;          // [KV, SEQ]   bf16 (V^T)
  unsigned short* ob = vt + (size_t)KV * SEQ;          // [SEQ, DIM]  bf16

  dim3 blk(256);
  // Q = x Wq^T
  gemm_wmma_kernel<false, true, false>
      <<<dim3(DIM / 128, SEQ / 128), blk, 0, stream>>>(x, Wq, qb, DIM, DIM);
  // K = x Wk^T
  gemm_wmma_kernel<false, true, false>
      <<<dim3(KV / 128, SEQ / 128), blk, 0, stream>>>(x, Wk, kb, DIM, KV);
  // V^T = (x Wv^T)^T  (stored transposed for P*V B-fragments)
  gemm_wmma_kernel<false, true, true>
      <<<dim3(KV / 128, SEQ / 128), blk, 0, stream>>>(x, Wv, vt, DIM, SEQ);
  // causal GQA flash attention
  flash_gqa_kernel<<<dim3(SEQ / 64, NH), dim3(128), 0, stream>>>(qb, kb, vt, ob);
  // out = attn Wo^T  (f32 output)
  gemm_wmma_kernel<true, false, false>
      <<<dim3(DIM / 128, SEQ / 128), blk, 0, stream>>>(ob, Wo, out, DIM, DIM);
}
